// MaskedAttention_32796370272780
// MI455X (gfx1250) — matrix operand order repaired and checked by reading, not compiled, not run on hardware
//
#include <hip/hip_runtime.h>
#include <cstdint>
#include <cstddef>

// ---------------------------------------------------------------------------
// MaskedAttention for MI455X (gfx1250): bf16 WMMA + TDM (tensor_load_to_lds)
// double-buffered A-tile staging. B=8, M=2048, D=1024.
// attention_mask is per-query-constant => softmax-invariant => ignored.
// ---------------------------------------------------------------------------

#define DEV __device__ __forceinline__

typedef __attribute__((ext_vector_type(16))) __bf16   v16bf;
typedef __attribute__((ext_vector_type(8)))  float    v8f;
typedef __attribute__((ext_vector_type(4)))  unsigned u32x4;
typedef __attribute__((ext_vector_type(4)))  unsigned su4;
typedef __attribute__((ext_vector_type(8)))  unsigned su8;

static constexpr int BATCH = 8;
static constexpr int SEQ   = 2048;
static constexpr int DIM   = 1024;
static constexpr int ROWS  = BATCH * SEQ;      // 16384
static constexpr float SCALE = 0.03125f;       // 1/sqrt(1024)

// A-tile staging: 32 rows x 128 K bf16 per chunk, TDM-padded 16B per 256B row
// -> LDS row pitch 136 elements (272B = 68 banks -> 4-bank skew per row,
//    conflict-free b128 fragment reads).
static constexpr int KCHUNK  = 128;
static constexpr int LDS_PITCH = 136;

DEV unsigned short f2bf(float f) {
  union { float f; unsigned u; } x; x.f = f;
  unsigned r = x.u + 0x7FFFu + ((x.u >> 16) & 1u);   // round-to-nearest-even
  return (unsigned short)(r >> 16);
}
DEV float bf2f(unsigned short h) {
  union { unsigned u; float f; } x; x.u = ((unsigned)h) << 16; return x.f;
}

union ABFrag { v16bf v; u32x4 q[2]; };

// A fragment (16x32 bf16) from LDS tile (row pitch LDS_PITCH).
// ISA layout: lanes 0-15 row M=lane hold K {0..7,16..23}; lanes 16-31 row
// M=lane-16 hold K {8..15,24..31}.
DEV v16bf load_a_frag_lds(const unsigned short* __restrict__ lds, int row_off,
                          int k0, int lane) {
  const int half = lane >> 4, r = lane & 15;
  const unsigned short* p = lds + (size_t)(row_off + r) * LDS_PITCH + k0 + 8 * half;
  ABFrag f;
  f.q[0] = *(const u32x4*)(p);
  f.q[1] = *(const u32x4*)(p + 16);
  return f.v;
}

// B fragment (32x16 bf16). Lane n = column; lanes 0-15 hold K {0..7,16..23}, 16-31 hold
// K {8..15,24..31}; source is row-major [n][k] (contiguous in k).
DEV v16bf load_b_frag(const unsigned short* __restrict__ Bptr, int ldb, int k0, int lane) {
  const int kh = lane >> 4, n = lane & 15;
  const unsigned short* p = Bptr + (size_t)n * ldb + k0 + 8 * kh;
  ABFrag f;
  f.q[0] = *(const u32x4*)(p);
  f.q[1] = *(const u32x4*)(p + 16);
  return f.v;
}

// Issue TDM: DMA a 2D bf16 tile (KCHUNK x 32 rows) global -> LDS with row padding.
// All descriptor words are wave-uniform (readfirstlane-forced into SGPRs).
DEV void tdm_load_a_tile(unsigned lds_addr, const unsigned short* gptr,
                         unsigned row_stride_elems) {
  const unsigned long long ga = (unsigned long long)(uintptr_t)gptr;
  su4 g0;
  g0[0] = __builtin_amdgcn_readfirstlane(1u);                       // count=1
  g0[1] = __builtin_amdgcn_readfirstlane(lds_addr);                 // lds_addr
  g0[2] = __builtin_amdgcn_readfirstlane((unsigned)(ga & 0xFFFFFFFFu));
  g0[3] = __builtin_amdgcn_readfirstlane(
            (unsigned)((ga >> 32) & 0x01FFFFFFu) | (2u << 30));     // addr[56:32] | type=2
  su8 g1;
  // data_size=1 (2B), pad_enable=1, pad_interval=5 (256B), pad_amount=3 (16B)
  g1[0] = __builtin_amdgcn_readfirstlane((1u << 16) | (1u << 20) | (5u << 22) | (3u << 25));
  g1[1] = __builtin_amdgcn_readfirstlane(((unsigned)KCHUNK & 0xFFFFu) << 16); // tensor_dim0 lo
  g1[2] = __builtin_amdgcn_readfirstlane(((unsigned)KCHUNK >> 16) | (32u << 16)); // dim0 hi | tensor_dim1 lo
  g1[3] = __builtin_amdgcn_readfirstlane(0u | ((unsigned)KCHUNK << 16));      // dim1 hi | tile_dim0
  g1[4] = __builtin_amdgcn_readfirstlane(32u);                                // tile_dim1 (tile_dim2=0)
  g1[5] = __builtin_amdgcn_readfirstlane(row_stride_elems);                   // tensor_dim0_stride[31:0]
  g1[6] = __builtin_amdgcn_readfirstlane(0u);
  g1[7] = __builtin_amdgcn_readfirstlane(0u);
  asm volatile("tensor_load_to_lds %0, %1" :: "s"(g0), "s"(g1) : "memory");
}

// ---------------------------------------------------------------------------
// Batched GEMM: C[b, r, n] = scale * sum_k A[b,r,k]*Bm[b,n,k] (+bias) (+epilogue)
//   mode 0: store f32   mode 1: store bf16   mode 2: ctx epilogue (bf16)
// Block = 128 threads (4 waves): 32 rows x 256 cols; wave tile 32 x 64
// (8 accumulators). A tile TDM-staged in LDS, double buffered, shared by all
// 4 waves; B fragments straight from global (distinct per wave, L2-resident).
// ---------------------------------------------------------------------------
__global__ __launch_bounds__(128)
void gemm_bf16_kernel(const unsigned short* __restrict__ A, int lda, long long sA,
                      const unsigned short* __restrict__ Bm, int ldb, long long sB,
                      void* __restrict__ C, int ldc, long long sC,
                      int Kk, float scale, const float* __restrict__ bias, int mode,
                      const float* __restrict__ diag_p,
                      const unsigned short* __restrict__ uv,
                      const unsigned short* __restrict__ tv)
{
  __shared__ unsigned short bufA[2][32 * LDS_PITCH];

  const int bz   = blockIdx.z;
  const int wave = threadIdx.x >> 5;
  const int lane = threadIdx.x & 31;
  const int row0 = blockIdx.y * 32;
  const int n0   = blockIdx.x * 256 + wave * 64;

  const unsigned short* Ab = A  + (size_t)bz * sA + (size_t)row0 * lda;
  const unsigned short* B0 = Bm + (size_t)bz * sB + (size_t)(n0     ) * ldb;
  const unsigned short* B1 = Bm + (size_t)bz * sB + (size_t)(n0 + 16) * ldb;
  const unsigned short* B2 = Bm + (size_t)bz * sB + (size_t)(n0 + 32) * ldb;
  const unsigned short* B3 = Bm + (size_t)bz * sB + (size_t)(n0 + 48) * ldb;

  const unsigned ldsOff0 = (unsigned)(uintptr_t)(&bufA[0][0]);
  const unsigned ldsOff1 = (unsigned)(uintptr_t)(&bufA[1][0]);
  const int nkc = Kk / KCHUNK;

  // prologue: stage chunk 0
  if (wave == 0) tdm_load_a_tile(ldsOff0, Ab, (unsigned)lda);

  v8f acc[2][4] = {};
  for (int kc = 0; kc < nkc; ++kc) {
    __syncthreads();   // everyone done reading buf[(kc+1)&1] from iteration kc-1
    if (wave == 0) {
      if (kc + 1 < nkc) {
        tdm_load_a_tile((kc & 1) ? ldsOff0 : ldsOff1,
                        Ab + (size_t)(kc + 1) * KCHUNK, (unsigned)lda);
        __builtin_amdgcn_s_wait_tensorcnt(1);   // chunk kc landed
      } else {
        __builtin_amdgcn_s_wait_tensorcnt(0);
      }
    }
    __syncthreads();   // buf[kc&1] visible to all waves
    const unsigned short* As = bufA[kc & 1];
    const int kg = kc * KCHUNK;

#pragma unroll
    for (int ks = 0; ks < KCHUNK; ks += 32) {
      const v16bf aLo = load_a_frag_lds(As, 0,  ks, lane);
      const v16bf aHi = load_a_frag_lds(As, 16, ks, lane);
      const v16bf b0 = load_b_frag(B0, ldb, kg + ks, lane);
      const v16bf b1 = load_b_frag(B1, ldb, kg + ks, lane);
      const v16bf b2 = load_b_frag(B2, ldb, kg + ks, lane);
      const v16bf b3 = load_b_frag(B3, ldb, kg + ks, lane);
      acc[0][0] = __builtin_amdgcn_wmma_f32_16x16x32_bf16(false, aLo, false, b0, (short)0, acc[0][0], false, false);
      acc[0][1] = __builtin_amdgcn_wmma_f32_16x16x32_bf16(false, aLo, false, b1, (short)0, acc[0][1], false, false);
      acc[0][2] = __builtin_amdgcn_wmma_f32_16x16x32_bf16(false, aLo, false, b2, (short)0, acc[0][2], false, false);
      acc[0][3] = __builtin_amdgcn_wmma_f32_16x16x32_bf16(false, aLo, false, b3, (short)0, acc[0][3], false, false);
      acc[1][0] = __builtin_amdgcn_wmma_f32_16x16x32_bf16(false, aHi, false, b0, (short)0, acc[1][0], false, false);
      acc[1][1] = __builtin_amdgcn_wmma_f32_16x16x32_bf16(false, aHi, false, b1, (short)0, acc[1][1], false, false);
      acc[1][2] = __builtin_amdgcn_wmma_f32_16x16x32_bf16(false, aHi, false, b2, (short)0, acc[1][2], false, false);
      acc[1][3] = __builtin_amdgcn_wmma_f32_16x16x32_bf16(false, aHi, false, b3, (short)0, acc[1][3], false, false);
      __builtin_prefetch(B0 + (size_t)(lane & 15) * ldb + kg + ks + KCHUNK, 0, 0);
      __builtin_prefetch(B2 + (size_t)(lane & 15) * ldb + kg + ks + KCHUNK, 0, 0);
    }
  }

  // C layout: VGPR r, lane l -> row = tile_row + r + 8*(l>>4), col = tile_n + (l&15)
  const int cr = 8 * (lane >> 4);
  const int cn = lane & 15;
#pragma unroll
  for (int s = 0; s < 2; ++s) {
#pragma unroll
    for (int t = 0; t < 4; ++t) {
      const v8f a = acc[s][t];
      const int col = n0 + t * 16 + cn;
      const float bsv = bias ? bias[col] : 0.0f;
#pragma unroll
      for (int r = 0; r < 8; ++r) {
        const int row = row0 + s * 16 + cr + r;
        float v = a[r] * scale + bsv;
        const size_t idx = (size_t)bz * sC + (size_t)row * ldc + col;
        if (mode == 0) {
          ((float*)C)[idx] = v;
        } else if (mode == 1) {
          ((unsigned short*)C)[idx] = f2bf(v);
        } else {
          const size_t g = ((size_t)(bz * SEQ + row)) * DIM + col;
          v += diag_p[(size_t)bz * SEQ + row] * (bf2f(uv[g]) - bf2f(tv[g]));
          ((unsigned short*)C)[idx] = f2bf(v);
        }
      }
    }
  }
}

// ---------------------------------------------------------------------------
__global__ __launch_bounds__(256)
void f32_to_bf16_kernel(const float* __restrict__ in, unsigned short* __restrict__ out, size_t n) {
  size_t i = (size_t)blockIdx.x * blockDim.x + threadIdx.x;
  const size_t stride = (size_t)gridDim.x * blockDim.x;
  for (; i < n; i += stride) out[i] = f2bf(in[i]);
}

// scores[b,i,i] = scale * dot(q[b,i,:], uk[b,i,:])   (one wave per row)
__global__ __launch_bounds__(256)
void diag_scores_kernel(const unsigned short* __restrict__ q,
                        const unsigned short* __restrict__ uk,
                        float* __restrict__ scores, float scale) {
  const int row  = blockIdx.x * 8 + (threadIdx.x >> 5);
  const int lane = threadIdx.x & 31;
  const unsigned short* qp = q  + (size_t)row * DIM;
  const unsigned short* kp = uk + (size_t)row * DIM;
  float s = 0.f;
  for (int j = lane; j < DIM; j += 32) s += bf2f(qp[j]) * bf2f(kp[j]);
#pragma unroll
  for (int o = 16; o; o >>= 1) s += __shfl_xor(s, o, 32);
  if (lane == 0) {
    const int b = row >> 11, i = row & (SEQ - 1);
    scores[((size_t)b * SEQ + i) * SEQ + i] = s * scale;
  }
}

// Row softmax over 2048 keys; store bf16 probs + f32 diagonal prob.
__global__ __launch_bounds__(256)
void softmax_kernel(const float* __restrict__ scores,
                    unsigned short* __restrict__ probs,
                    float* __restrict__ diag_p) {
  __shared__ float sm[8];
  const int row = blockIdx.x, t = threadIdx.x;
  const float4* sp = (const float4*)(scores + (size_t)row * SEQ);
  const float4 v0 = sp[t * 2], v1 = sp[t * 2 + 1];
  float f[8] = {v0.x, v0.y, v0.z, v0.w, v1.x, v1.y, v1.z, v1.w};

  float mx = f[0];
#pragma unroll
  for (int u = 1; u < 8; ++u) mx = fmaxf(mx, f[u]);
#pragma unroll
  for (int o = 16; o; o >>= 1) mx = fmaxf(mx, __shfl_xor(mx, o, 32));
  if ((t & 31) == 0) sm[t >> 5] = mx;
  __syncthreads();
  float m = sm[0];
#pragma unroll
  for (int w = 1; w < 8; ++w) m = fmaxf(m, sm[w]);
  __syncthreads();

  float e[8], s = 0.f;
#pragma unroll
  for (int u = 0; u < 8; ++u) { e[u] = __expf(f[u] - m); s += e[u]; }
#pragma unroll
  for (int o = 16; o; o >>= 1) s += __shfl_xor(s, o, 32);
  if ((t & 31) == 0) sm[t >> 5] = s;
  __syncthreads();
  float l = 0.f;
#pragma unroll
  for (int w = 0; w < 8; ++w) l += sm[w];
  const float inv = 1.0f / l;

  unsigned short* pp = probs + (size_t)row * SEQ + t * 8;
#pragma unroll
  for (int u = 0; u < 8; ++u) pp[u] = f2bf(e[u] * inv);

  const int di = row & (SEQ - 1);
  if ((di >> 3) == t) {
#pragma unroll
    for (int u = 0; u < 8; ++u)
      if ((di & 7) == u) diag_p[row] = e[u] * inv;
  }
}

// LDS-tiled bf16 transpose: in [rows x cols] -> out [cols x rows], per batch z.
__global__ __launch_bounds__(256)
void transpose_bf16_kernel(const unsigned short* __restrict__ in,
                           unsigned short* __restrict__ out, int rows, int cols) {
  __shared__ unsigned short tile[32][33];
  const int b = blockIdx.z;
  const unsigned short* ib = in  + (size_t)b * rows * cols;
  unsigned short*       ob = out + (size_t)b * rows * cols;
  const int x0 = blockIdx.x * 32, y0 = blockIdx.y * 32;
  const int tx = threadIdx.x, ty = threadIdx.y;  // 32 x 8
#pragma unroll
  for (int k = 0; k < 32; k += 8) tile[ty + k][tx] = ib[(size_t)(y0 + ty + k) * cols + x0 + tx];
  __syncthreads();
#pragma unroll
  for (int k = 0; k < 32; k += 8) ob[(size_t)(x0 + ty + k) * rows + y0 + tx] = tile[tx][ty + k];
}

// LayerNorm over D=1024; 256 threads x float4 per row.
__global__ __launch_bounds__(256)
void layernorm_kernel(const float* __restrict__ in, const float* __restrict__ g,
                      const float* __restrict__ bta, float* __restrict__ out) {
  __shared__ float sm[8];
  const int row = blockIdx.x, t = threadIdx.x;
  const float4 v = ((const float4*)(in + (size_t)row * DIM))[t];

  float s = v.x + v.y + v.z + v.w;
#pragma unroll
  for (int o = 16; o; o >>= 1) s += __shfl_xor(s, o, 32);
  if ((t & 31) == 0) sm[t >> 5] = s;
  __syncthreads();
  float tot = 0.f;
#pragma unroll
  for (int w = 0; w < 8; ++w) tot += sm[w];
  __syncthreads();
  const float mu = tot * (1.0f / DIM);

  const float dx = v.x - mu, dy = v.y - mu, dz = v.z - mu, dw = v.w - mu;
  float vs = dx * dx + dy * dy + dz * dz + dw * dw;
#pragma unroll
  for (int o = 16; o; o >>= 1) vs += __shfl_xor(vs, o, 32);
  if ((t & 31) == 0) sm[t >> 5] = vs;
  __syncthreads();
  float vtot = 0.f;
#pragma unroll
  for (int w = 0; w < 8; ++w) vtot += sm[w];
  const float rstd = rsqrtf(vtot * (1.0f / DIM) + 1e-12f);

  const float4 gv = ((const float4*)g)[t];
  const float4 bv = ((const float4*)bta)[t];
  float4 o;
  o.x = dx * rstd * gv.x + bv.x;
  o.y = dy * rstd * gv.y + bv.y;
  o.z = dz * rstd * gv.z + bv.z;
  o.w = dw * rstd * gv.w + bv.w;
  ((float4*)(out + (size_t)row * DIM))[t] = o;
}

// ---------------------------------------------------------------------------
extern "C" void kernel_launch(void* const* d_in, const int* in_sizes, int n_in,
                              void* d_out, int out_size, void* d_ws, size_t ws_size,
                              hipStream_t stream) {
  (void)in_sizes; (void)n_in; (void)out_size; (void)ws_size;

  const float* hsT = (const float*)d_in[0];
  const float* hsU = (const float*)d_in[1];
  // d_in[2] = attention_mask: per-query additive constant -> softmax invariant -> unused.
  const float* Wq = (const float*)d_in[3];  const float* bq = (const float*)d_in[4];
  const float* Wk = (const float*)d_in[5];  const float* bk = (const float*)d_in[6];
  const float* Wv = (const float*)d_in[7];  const float* bv = (const float*)d_in[8];
  const float* Wo = (const float*)d_in[9];  const float* bo = (const float*)d_in[10];
  const float* ln_g = (const float*)d_in[11];
  const float* ln_b = (const float*)d_in[12];
  float* out = (float*)d_out;

  // ---- workspace carve-up (256B aligned) ----
  char* ws = (char*)d_ws;
  size_t off = 0;
  auto alloc = [&](size_t bytes) { size_t r = off; off += (bytes + 255) & ~(size_t)255; return r; };
  const size_t szHS  = (size_t)ROWS * DIM * 2;       // bf16 [16384 x 1024]
  const size_t szW   = (size_t)DIM * DIM * 2;        // bf16 [1024 x 1024]
  const size_t szSco = (size_t)BATCH * SEQ * SEQ * 4;
  const size_t szPrb = (size_t)BATCH * SEQ * SEQ * 2;

  unsigned short* hsT_bf = (unsigned short*)(ws + alloc(szHS));
  unsigned short* hsU_bf = (unsigned short*)(ws + alloc(szHS));
  unsigned short* Wq_bf  = (unsigned short*)(ws + alloc(szW));
  unsigned short* Wk_bf  = (unsigned short*)(ws + alloc(szW));
  unsigned short* Wv_bf  = (unsigned short*)(ws + alloc(szW));
  unsigned short* Wo_bf  = (unsigned short*)(ws + alloc(szW));
  unsigned short* q_bf   = (unsigned short*)(ws + alloc(szHS));
  unsigned short* tk_bf  = (unsigned short*)(ws + alloc(szHS));
  unsigned short* uk_bf  = (unsigned short*)(ws + alloc(szHS));
  unsigned short* uv_bf  = (unsigned short*)(ws + alloc(szHS));
  unsigned short* tv_bf  = (unsigned short*)(ws + alloc(szHS));
  float*          scores = (float*)(ws + alloc(szSco));     // later reused as out_pre f32
  unsigned short* probs  = (unsigned short*)(ws + alloc(szPrb));
  float*          diag_p = (float*)(ws + alloc((size_t)ROWS * 4));
  unsigned short* tvT_bf = (unsigned short*)(ws + alloc(szHS));
  unsigned short* ctx_bf = (unsigned short*)(ws + alloc(szHS));
  float*          out_pre = scores;                         // 67MB fits in 134MB region

  // ---- 1) f32 -> bf16 conversions ----
  auto cvt = [&](const float* src, unsigned short* dst, size_t n) {
    f32_to_bf16_kernel<<<4096, 256, 0, stream>>>(src, dst, n);
  };
  cvt(hsT, hsT_bf, (size_t)ROWS * DIM);
  cvt(hsU, hsU_bf, (size_t)ROWS * DIM);
  cvt(Wq, Wq_bf, (size_t)DIM * DIM);
  cvt(Wk, Wk_bf, (size_t)DIM * DIM);
  cvt(Wv, Wv_bf, (size_t)DIM * DIM);
  cvt(Wo, Wo_bf, (size_t)DIM * DIM);

  // ---- generic batched GEMM launcher (32-row, 256-col blocks) ----
  auto gemm = [&](const unsigned short* A, int lda, long long sA,
                  const unsigned short* Bm, int ldb, long long sB,
                  void* C, int ldc, long long sC,
                  int R, int N, int K, float scale, const float* bias, int mode,
                  const float* dp, const unsigned short* uvp, const unsigned short* tvp, int nb) {
    dim3 grid(N / 256, R / 32, nb), block(128);
    gemm_bf16_kernel<<<grid, block, 0, stream>>>(A, lda, sA, Bm, ldb, sB, C, ldc, sC,
                                                 K, scale, bias, mode, dp, uvp, tvp);
  };

  // ---- 2) five input projections: X @ W^T + b  (bf16 out) ----
  gemm(hsU_bf, DIM, 0, Wq_bf, DIM, 0, q_bf,  DIM, 0, ROWS, DIM, DIM, 1.f, bq, 1, nullptr, nullptr, nullptr, 1);
  gemm(hsT_bf, DIM, 0, Wk_bf, DIM, 0, tk_bf, DIM, 0, ROWS, DIM, DIM, 1.f, bk, 1, nullptr, nullptr, nullptr, 1);
  gemm(hsU_bf, DIM, 0, Wk_bf, DIM, 0, uk_bf, DIM, 0, ROWS, DIM, DIM, 1.f, bk, 1, nullptr, nullptr, nullptr, 1);
  gemm(hsU_bf, DIM, 0, Wv_bf, DIM, 0, uv_bf, DIM, 0, ROWS, DIM, DIM, 1.f, bv, 1, nullptr, nullptr, nullptr, 1);
  gemm(hsT_bf, DIM, 0, Wv_bf, DIM, 0, tv_bf, DIM, 0, ROWS, DIM, DIM, 1.f, bv, 1, nullptr, nullptr, nullptr, 1);

  // ---- 3) scores = scale * Q @ TK^T  (batched, f32 out) ----
  gemm(q_bf, DIM, (long long)SEQ * DIM, tk_bf, DIM, (long long)SEQ * DIM,
       scores, SEQ, (long long)SEQ * SEQ, SEQ, SEQ, DIM, SCALE, nullptr, 0,
       nullptr, nullptr, nullptr, BATCH);

  // ---- 4) diagonal override: scores[b,i,i] = scale * q.uk ----
  diag_scores_kernel<<<ROWS / 8, 256, 0, stream>>>(q_bf, uk_bf, scores, SCALE);

  // ---- 5) softmax -> bf16 probs + f32 diagonal prob ----
  softmax_kernel<<<ROWS, 256, 0, stream>>>(scores, probs, diag_p);

  // ---- 6) transpose tv -> tvT [b, d, j] so ctx-GEMM B operand is lane-contiguous ----
  {
    dim3 grid(DIM / 32, SEQ / 32, BATCH), block(32, 8);
    transpose_bf16_kernel<<<grid, block, 0, stream>>>(tv_bf, tvT_bf, SEQ, DIM);
  }

  // ---- 7) ctx = P @ TV, + p_ii*(uv - tv) epilogue, bf16 out ----
  gemm(probs, SEQ, (long long)SEQ * SEQ, tvT_bf, SEQ, (long long)DIM * SEQ,
       ctx_bf, DIM, (long long)SEQ * DIM, SEQ, DIM, SEQ, 1.f, nullptr, 2,
       diag_p, uv_bf, tv_bf, BATCH);

  // ---- 8) out_pre = ctx @ Wo^T + bo  (f32 out, reuses scores region) ----
  gemm(ctx_bf, DIM, 0, Wo_bf, DIM, 0, out_pre, DIM, 0, ROWS, DIM, DIM, 1.f, bo, 0,
       nullptr, nullptr, nullptr, 1);

  // ---- 9) LayerNorm -> d_out ----
  layernorm_kernel<<<ROWS, 256, 0, stream>>>(out_pre, ln_g, ln_b, out);
}
